// BiMambaEncoder_24790551233471
// MI455X (gfx1250) — matrix operand order repaired and checked by reading, not compiled, not run on hardware
//
#include <hip/hip_runtime.h>
#include <math.h>

// ---------------------------------------------------------------------------
// Types for CDNA5 WMMA (gfx1250, wave32)
// ---------------------------------------------------------------------------
typedef __bf16 bf16;
typedef __attribute__((ext_vector_type(8)))  __bf16 v8bf;
typedef __attribute__((ext_vector_type(16))) __bf16 v16bf;
typedef __attribute__((ext_vector_type(8)))  float  v8f;
typedef __attribute__((ext_vector_type(4)))  int    i32x4;

#define Lq   2048
#define Bq   4
#define DIq  1024
#define DSq  16
#define Mq   (Bq * Lq)   // 8192 tokens

// Async memory->LDS path (CDNA5 GLOBAL_LOAD_ASYNC_TO_LDS, tracked by ASYNCcnt)
#if __has_builtin(__builtin_amdgcn_global_load_async_to_lds_b128) && \
    __has_builtin(__builtin_amdgcn_s_wait_asynccnt)
#define HAVE_ASYNC_LDS 1
#define AS_GLOBAL __attribute__((address_space(1)))
#define AS_LOCAL  __attribute__((address_space(3)))
#else
#define HAVE_ASYNC_LDS 0
#endif

// ---------------------------------------------------------------------------
// f32 -> bf16 conversion kernel (weights + activations)
// ---------------------------------------------------------------------------
__global__ void cvt_bf16_kernel(const float* __restrict__ in, bf16* __restrict__ out, size_t n) {
    size_t i = (size_t)blockIdx.x * blockDim.x + threadIdx.x;
    if (i < n) out[i] = (bf16)in[i];
}

// ---------------------------------------------------------------------------
// WMMA GEMM: D[M,N] = A[M,K] * W[N,K]^T (+bias) (+softplus)
//   A bf16 row-major (leading dim lda), W bf16 row-major [N,K] (ldb == K).
//   Each wave computes a 32x32 tile: 4x v_wmma_f32_16x16x32_bf16 per K-step.
//   Block = 256 threads = 8 waves laid out 4 (M) x 2 (N) -> 128x64 block tile.
// ---------------------------------------------------------------------------
__device__ __forceinline__ v16bf load_fragA(const bf16* base, int ld, int row, int k0, int khalf) {
    // 16-bit A 16x32 layout: lanes 0-15 hold K {0..7, 16..23}; lanes 16-31 {8..15, 24..31}
    const bf16* p = base + (size_t)row * ld + k0 + khalf * 8;
    v8bf lo = *(const v8bf*)p;
    v8bf hi = *(const v8bf*)(p + 16);
    return __builtin_shufflevector(lo, hi, 0,1,2,3,4,5,6,7,8,9,10,11,12,13,14,15);
}

__device__ __forceinline__ v16bf load_fragB(const bf16* base, int ld, int row, int k0, int khalf) {
    // 16-bit B 32x16 layout: lanes 0-15 hold K {0..7, 16..23}; lanes 16-31 {8..15, 24..31}
    const bf16* p = base + (size_t)row * ld + k0 + khalf * 8;
    v8bf lo = *(const v8bf*)p;
    v8bf hi = *(const v8bf*)(p + 16);
    return __builtin_shufflevector(lo, hi, 0,1,2,3,4,5,6,7,8,9,10,11,12,13,14,15);
}

__device__ __forceinline__ void store_tile(v8f c, int mbase, int nbase, int l16, int khalf,
                                           const float* __restrict__ bias,
                                           float* __restrict__ outF, bf16* __restrict__ outB,
                                           int N, int softplus) {
    int col = nbase + l16;
    float bv = bias ? bias[col] : 0.0f;
#pragma unroll
    for (int r = 0; r < 8; ++r) {
        int row = mbase + r + 8 * khalf;
        float v = c[r] + bv;
        if (softplus) v = (v > 20.0f) ? v : log1pf(__expf(v));
        size_t off = (size_t)row * N + col;
        if (outF) outF[off] = v;
        if (outB) outB[off] = (bf16)v;
    }
}

__global__ __launch_bounds__(256) void wmma_gemm_kernel(
    const bf16* __restrict__ A, const bf16* __restrict__ W,
    const float* __restrict__ bias,
    float* __restrict__ outF, bf16* __restrict__ outB,
    int M, int N, int K, int lda, int softplus) {
    int lane  = threadIdx.x & 31;
    int wave  = threadIdx.x >> 5;
    int wm    = wave & 3;
    int wn    = wave >> 2;
    int m0    = blockIdx.y * 128 + wm * 32;
    int n0    = blockIdx.x * 64  + wn * 32;
    int l16   = lane & 15;
    int khalf = lane >> 4;
    int ra    = m0 + l16;
    int rb    = n0 + l16;

    v8f c00 = {}, c01 = {}, c10 = {}, c11 = {};
    for (int k0 = 0; k0 < K; k0 += 32) {
        if (k0 + 32 < K) {
            // near-scope (WGP) prefetch of next K-block
            __builtin_prefetch(A + (size_t)ra * lda + k0 + 32, 0, 3);
            __builtin_prefetch(W + (size_t)rb * K   + k0 + 32, 0, 3);
        }
        v16bf a0 = load_fragA(A, lda, ra,      k0, khalf);
        v16bf a1 = load_fragA(A, lda, ra + 16, k0, khalf);
        v16bf b0 = load_fragB(W, K,   rb,      k0, khalf);
        v16bf b1 = load_fragB(W, K,   rb + 16, k0, khalf);
        c00 = __builtin_amdgcn_wmma_f32_16x16x32_bf16(false, a0, false, b0, (short)0, c00, false, false);
        c01 = __builtin_amdgcn_wmma_f32_16x16x32_bf16(false, a0, false, b1, (short)0, c01, false, false);
        c10 = __builtin_amdgcn_wmma_f32_16x16x32_bf16(false, a1, false, b0, (short)0, c10, false, false);
        c11 = __builtin_amdgcn_wmma_f32_16x16x32_bf16(false, a1, false, b1, (short)0, c11, false, false);
    }
    store_tile(c00, m0,      n0,      l16, khalf, bias, outF, outB, N, softplus);
    store_tile(c01, m0,      n0 + 16, l16, khalf, bias, outF, outB, N, softplus);
    store_tile(c10, m0 + 16, n0,      l16, khalf, bias, outF, outB, N, softplus);
    store_tile(c11, m0 + 16, n0 + 16, l16, khalf, bias, outF, outB, N, softplus);
}

// ---------------------------------------------------------------------------
// Causal depthwise conv1d (K=4) + bias + SiLU. Direction handled by index map:
// scan-time index i maps to token l = i (fwd) or L-1-i (bwd); outputs written
// back at the mapped position so the final "flip back" is free.
// ---------------------------------------------------------------------------
__global__ void conv_silu_kernel(const float* __restrict__ xz,   // [M, 2048], u = cols 0..1023
                                 const float* __restrict__ cw,   // [DI,1,4]
                                 const float* __restrict__ cb,   // [DI]
                                 float* __restrict__ uc, bf16* __restrict__ ubf, int dir) {
    int d = blockIdx.x * blockDim.x + threadIdx.x;   // 0..1023
    int i = blockIdx.y;                              // scan index
    int b = blockIdx.z;
    int t = b * Lq + (dir ? (Lq - 1 - i) : i);
    float acc = cb[d];
#pragma unroll
    for (int j = 0; j < 4; ++j) {
        int ii = i - 3 + j;
        if (ii >= 0) {
            int tt = b * Lq + (dir ? (Lq - 1 - ii) : ii);
            acc += cw[d * 4 + j] * xz[(size_t)tt * 2048 + d];
        }
    }
    float s = acc / (1.0f + __expf(-acc));           // SiLU
    uc[(size_t)t * DIq + d]  = s;
    ubf[(size_t)t * DIq + d] = (bf16)s;
}

// ---------------------------------------------------------------------------
// Selective scan: one thread per (b, d) channel, DS=16 state in registers.
// Per-token B/C vectors (shared by all d) staged through LDS in 16-step
// chunks using CDNA5 async memory->LDS transfers (ASYNCcnt) when available.
// ---------------------------------------------------------------------------
__global__ __launch_bounds__(1024) void scan_kernel(
    const float* __restrict__ uc,    // [M, DI]
    const float* __restrict__ delta, // [M, DI]
    const float* __restrict__ xdb,   // [M, 64]: [dt32 | B16 | C16]
    const float* __restrict__ A_log, // [DI, DS]
    float* __restrict__ ys, int dir) {
    int d = threadIdx.x;
    int b = blockIdx.x;
    float Ar[DSq], h[DSq];
#pragma unroll
    for (int n = 0; n < DSq; ++n) { Ar[n] = -__expf(A_log[d * DSq + n]); h[n] = 0.0f; }

    __shared__ float sBC[16][32];    // [row][0..15]=B, [row][16..31]=C

    for (int ii = 0; ii < Lq; ii += 16) {
#if HAVE_ASYNC_LDS
        if (d < 128) {               // waves 0..3, wave-uniform predicate
            int row = d >> 3, part = d & 7;      // 16 rows x 8 x b128
            int i = ii + row;
            int t = b * Lq + (dir ? (Lq - 1 - i) : i);
            const float* g = xdb + (size_t)t * 64 + 32 + part * 4;
            float* l = &sBC[row][part * 4];
            __builtin_amdgcn_global_load_async_to_lds_b128(
                (AS_GLOBAL i32x4*)g, (AS_LOCAL i32x4*)l, 0, 0);
            __builtin_amdgcn_s_wait_asynccnt(0);
        }
        __syncthreads();
#else
        if (d < 512) {               // 16 rows x 32 cols cooperative load
            int row = d >> 5, c = d & 31;
            int i = ii + row;
            int t = b * Lq + (dir ? (Lq - 1 - i) : i);
            sBC[row][c] = xdb[(size_t)t * 64 + 32 + c];
        }
        __syncthreads();
#endif
#pragma unroll 1
        for (int s = 0; s < 16; ++s) {
            int i = ii + s;
            int t = b * Lq + (dir ? (Lq - 1 - i) : i);
            float dt = delta[(size_t)t * DIq + d];
            float ut = uc[(size_t)t * DIq + d];
            float du = dt * ut;
            float y = 0.0f;
#pragma unroll
            for (int n = 0; n < DSq; ++n) {
                h[n] = __expf(dt * Ar[n]) * h[n] + du * sBC[s][n];
                y += h[n] * sBC[s][16 + n];
            }
            ys[(size_t)t * DIq + d] = y;
        }
        __syncthreads();
    }
}

// ---------------------------------------------------------------------------
// y = (ys + uc*D) * silu(z)  -> bf16 for out_proj GEMM
// ---------------------------------------------------------------------------
__global__ void fuse_gate_kernel(const float* __restrict__ ys, const float* __restrict__ uc,
                                 const float* __restrict__ xz, const float* __restrict__ Dw,
                                 bf16* __restrict__ ybf) {
    size_t idx = (size_t)blockIdx.x * blockDim.x + threadIdx.x;  // over M*DI
    int d = (int)(idx & (DIq - 1));
    size_t t = idx >> 10;
    float y = ys[idx] + uc[idx] * Dw[d];
    float z = xz[t * 2048 + 1024 + d];
    float g = z / (1.0f + __expf(-z));
    ybf[idx] = (bf16)(y * g);
}

// ---------------------------------------------------------------------------
// Concat [xf|xb], RMSNorm over 1024 dims, scale, -> bf16
// ---------------------------------------------------------------------------
__global__ __launch_bounds__(256) void rmsnorm_kernel(const float* __restrict__ xf,
                                                      const float* __restrict__ xb,
                                                      const float* __restrict__ nw,
                                                      bf16* __restrict__ out) {
    int t = blockIdx.x;
    __shared__ float red[256];
    float v[4];
    float ss = 0.0f;
#pragma unroll
    for (int q = 0; q < 4; ++q) {
        int col = threadIdx.x + q * 256;
        float x = (col < 512) ? xf[(size_t)t * 512 + col] : xb[(size_t)t * 512 + col - 512];
        v[q] = x;
        ss += x * x;
    }
    red[threadIdx.x] = ss;
    __syncthreads();
    for (int s = 128; s > 0; s >>= 1) {
        if (threadIdx.x < s) red[threadIdx.x] += red[threadIdx.x + s];
        __syncthreads();
    }
    float r = rsqrtf(red[0] * (1.0f / 1024.0f) + 1e-5f);
#pragma unroll
    for (int q = 0; q < 4; ++q) {
        int col = threadIdx.x + q * 256;
        out[(size_t)t * 1024 + col] = (bf16)(v[q] * r * nw[col]);
    }
}

// ---------------------------------------------------------------------------
// out[b, j] = tanh(max_l feat[b, l, j])
// ---------------------------------------------------------------------------
__global__ void maxtanh_kernel(const float* __restrict__ feat, float* __restrict__ out) {
    int j = blockIdx.x * blockDim.x + threadIdx.x;  // 0..511
    int b = blockIdx.y;
    float m = -3.4e38f;
    for (int l = 0; l < Lq; ++l) {
        m = fmaxf(m, feat[((size_t)b * Lq + l) * 512 + j]);
    }
    out[b * 512 + j] = tanhf(m);
}

// ---------------------------------------------------------------------------
// Host orchestration
// ---------------------------------------------------------------------------
extern "C" void kernel_launch(void* const* d_in, const int* in_sizes, int n_in,
                              void* d_out, int out_size, void* d_ws, size_t ws_size,
                              hipStream_t stream) {
    (void)in_sizes; (void)n_in; (void)out_size; (void)ws_size;
    const size_t M = Mq;

    size_t off = 0;
    auto take = [&](size_t bytes) -> void* {
        void* p = (char*)d_ws + off;
        off += (bytes + 255) & ~(size_t)255;
        return p;
    };
    bf16*  XBF   = (bf16*) take(M * 768  * 2);
    bf16*  HBF   = (bf16*) take(M * 512  * 2);
    float* XF    = (float*)take(M * 512  * 4);
    float* XB    = (float*)take(M * 512  * 4);
    bf16*  XBIBF = (bf16*) take(M * 1024 * 2);
    float* FEAT  = (float*)take(M * 512  * 4);
    float* XZ    = (float*)take(M * 2048 * 4);
    float* UC    = (float*)take(M * 1024 * 4);
    bf16*  UBF   = (bf16*) take(M * 1024 * 2);
    float* XDB   = (float*)take(M * 64   * 4);
    bf16*  XDBBF = (bf16*) take(M * 64   * 2);
    float* DELTA = (float*)take(M * 1024 * 4);
    float* YS    = (float*)take(M * 1024 * 4);
    bf16*  YBF   = (bf16*) take(M * 1024 * 2);
    bf16*  Wpin  = (bf16*) take(512  * 768 * 2);
    bf16*  Wpout = (bf16*) take(512  * 1024 * 2);
    bf16*  Winp  = (bf16*) take(2048 * 512 * 2);
    bf16*  Wxp   = (bf16*) take(64   * 1024 * 2);
    bf16*  Wdt   = (bf16*) take(1024 * 32 * 2);
    bf16*  Wout  = (bf16*) take(512  * 1024 * 2);

    auto cvt = [&](const float* src, bf16* dst, size_t n) {
        cvt_bf16_kernel<<<dim3((unsigned)((n + 255) / 256)), 256, 0, stream>>>(src, dst, n);
    };
    auto gemm = [&](const bf16* A, const bf16* W, const float* bias,
                    float* oF, bf16* oB, int Mg, int Ng, int Kg, int lda, int sp) {
        dim3 grid(Ng / 64, Mg / 128);
        wmma_gemm_kernel<<<grid, 256, 0, stream>>>(A, W, bias, oF, oB, Mg, Ng, Kg, lda, sp);
    };

    const float* x         = (const float*)d_in[0];
    const float* proj_in_w = (const float*)d_in[1];
    const float* proj_in_b = (const float*)d_in[2];
    const float* norm_w    = (const float*)d_in[21];
    const float* proj_ow   = (const float*)d_in[22];
    const float* proj_ob   = (const float*)d_in[23];

    // h = x @ proj_in_w.T + proj_in_b  (bf16 output feeds both directions)
    cvt(x, XBF, M * 768);
    cvt(proj_in_w, Wpin, 512 * 768);
    gemm(XBF, Wpin, proj_in_b, nullptr, HBF, (int)M, 512, 768, 768, 0);
    cvt(proj_ow, Wpout, 512 * 1024);

    for (int dir = 0; dir < 2; ++dir) {
        int base = 3 + dir * 9;
        const float* in_proj_w = (const float*)d_in[base + 0];
        const float* conv_w    = (const float*)d_in[base + 1];
        const float* conv_b    = (const float*)d_in[base + 2];
        const float* x_proj_w  = (const float*)d_in[base + 3];
        const float* dt_proj_w = (const float*)d_in[base + 4];
        const float* dt_proj_b = (const float*)d_in[base + 5];
        const float* A_log     = (const float*)d_in[base + 6];
        const float* Dw        = (const float*)d_in[base + 7];
        const float* out_pw    = (const float*)d_in[base + 8];

        // xz = h @ in_proj_w.T   [M, 2048] f32
        cvt(in_proj_w, Winp, 2048 * 512);
        gemm(HBF, Winp, nullptr, XZ, nullptr, (int)M, 2048, 512, 512, 0);

        // depthwise conv + SiLU (direction-aware)
        conv_silu_kernel<<<dim3(DIq / 256, Lq, Bq), 256, 0, stream>>>(XZ, conv_w, conv_b, UC, UBF, dir);

        // xdb = u @ x_proj_w.T   [M, 64] (f32 + bf16 for the dt GEMM)
        cvt(x_proj_w, Wxp, 64 * 1024);
        gemm(UBF, Wxp, nullptr, XDB, XDBBF, (int)M, 64, 1024, 1024, 0);

        // delta = softplus(dt @ dt_proj_w.T + b)   K=32 -> one WMMA per tile
        cvt(dt_proj_w, Wdt, 1024 * 32);
        gemm(XDBBF, Wdt, dt_proj_b, DELTA, nullptr, (int)M, 1024, 32, 64, 1);

        // sequential selective scan (direction-aware)
        scan_kernel<<<dim3(Bq), 1024, 0, stream>>>(UC, DELTA, XDB, A_log, YS, dir);

        // y = (ys + u*D) * silu(z) -> bf16
        fuse_gate_kernel<<<dim3((unsigned)(M * 1024 / 256)), 256, 0, stream>>>(YS, UC, XZ, Dw, YBF);

        // out = y @ out_proj_w.T
        cvt(out_pw, Wout, 512 * 1024);
        gemm(YBF, Wout, nullptr, dir ? XB : XF, (bf16*)nullptr, (int)M, 512, 1024, 1024, 0);
    }

    // RMSNorm over concat -> bf16
    rmsnorm_kernel<<<dim3((unsigned)M), 256, 0, stream>>>(XF, XB, norm_w, XBIBF);

    // feat = xbi @ proj_out_w.T + b
    gemm(XBIBF, Wpout, proj_ob, FEAT, nullptr, (int)M, 512, 1024, 1024, 0);

    // out = tanh(max over L)
    maxtanh_kernel<<<dim3(2, Bq), 256, 0, stream>>>(FEAT, (float*)d_out);
}
